// GNNBaseLayer_64957085384837
// MI455X (gfx1250) — matrix operand order repaired and checked by reading, not compiled, not run on hardware
//
#include <hip/hip_runtime.h>

#define NNODES 50000
#define NEDGES 800000
#define EDGE_BLOCKS 1250   // persistent blocks; each handles (E/64)/1250 = 10 chunks
#define NODE_BLOCKS 250

typedef __attribute__((ext_vector_type(16))) __bf16 v16bf;
typedef __attribute__((ext_vector_type(8)))  float  v8f;

union ABf { v16bf h; unsigned int u[8]; };

// Native f32->bf16 (v_cvt_pk_bf16_f32 when paired).
__device__ __forceinline__ unsigned short bfc(float f) {
  return __builtin_bit_cast(unsigned short, (__bf16)f);
}
__device__ __forceinline__ unsigned int pk2(float lo, float hi) {
  return (unsigned int)bfc(lo) | ((unsigned int)bfc(hi) << 16);
}

__device__ __forceinline__ float fast_tanh(float x) {
#if __has_builtin(__builtin_amdgcn_tanhf)
  return __builtin_amdgcn_tanhf(x);   // v_tanh_f32 (TRANS, co-executes with VALU)
#else
  return tanhf(x);
#endif
}

__device__ __forceinline__ float gelu_f(float x) {
  float u = x * (0.7978845608028654f + 0.035677408136300125f * x * x);
  return 0.5f * x * (1.0f + fast_tanh(u));
}

__device__ __forceinline__ void atomic_add_f32(float* p, float v) {
  unsafeAtomicAdd(p, v);  // native global_atomic_add_f32
}

__device__ __forceinline__ v8f wmma_bf16(ABf a, ABf b, v8f c) {
  return __builtin_amdgcn_wmma_f32_16x16x32_bf16(
      /*neg_a=*/false, a.h, /*neg_b=*/false, b.h,
      /*c_mod=*/(short)0, c, /*reuse_a=*/false, /*reuse_b=*/false);
}

// Build A fragment (16xK tile, 16-bit A layout) from LDS staging tile.
// stage is dword-addressed, rowDw dwords per row (K/2). kbase in {0,32,64,96}.
__device__ __forceinline__ ABf load_a(const unsigned int* st, int rowDw, int kbase, int lane) {
  ABf a;
  const int m  = lane & 15;
  const int hb = (lane >= 16) ? 8 : 0;
#pragma unroll
  for (int v = 0; v < 8; ++v) {
    int kk = kbase + 2 * hb + (v << 1);
    a.u[v] = st[m * rowDw + (kk >> 1)];
  }
  return a;
}

// Build B fragment from transposed weights in LDS: WT[n][k] bf16, rowDw dwords/row.
__device__ __forceinline__ ABf load_b(const unsigned int* wt, int rowDw, int ncol, int kbase, int lane) {
  ABf b;
  int base = ncol * rowDw + (kbase >> 1) + ((lane >= 16) ? 8 : 0);
#pragma unroll
  for (int v = 0; v < 8; ++v) b.u[v] = wt[base + v];
  return b;
}

__global__ void zero_ws_kernel(float* __restrict__ p, long n) {
  long i = (long)blockIdx.x * blockDim.x + threadIdx.x;
  if (i < n) p[i] = 0.0f;
}

// ---------------- Edge pipeline: gather+BN -> GEMM -> GEMM, edge MLP, gate, scatter ----------------
__global__ __launch_bounds__(128) void edge_kernel(
    const float* __restrict__ node_feats, const float* __restrict__ edge_feats,
    const int* __restrict__ src, const int* __restrict__ dst,
    const float* __restrict__ m_g1, const float* __restrict__ m_b1,
    const float* __restrict__ m_mu1, const float* __restrict__ m_v1,
    const float* __restrict__ m_W1, const float* __restrict__ m_c1,
    const float* __restrict__ m_g2, const float* __restrict__ m_b2,
    const float* __restrict__ m_mu2, const float* __restrict__ m_v2,
    const float* __restrict__ m_W2, const float* __restrict__ m_c2,
    const float* __restrict__ e_W1, const float* __restrict__ e_c1,
    const float* __restrict__ e_W2, const float* __restrict__ e_c2,
    float* __restrict__ seg_sum, float* __restrict__ seg_cnt) {
  __shared__ unsigned int sWm1[64 * 32];
  __shared__ unsigned int sWm2[64 * 32];
  __shared__ unsigned int sWe1[64 * 32];
  __shared__ unsigned int sWe2[64 * 32];
  __shared__ float sSc1[64], sSh1[64], sSc2[64], sSh2[64];
  __shared__ float sCm1[64], sCm2[64], sCe1[64], sCe2[64];
  __shared__ unsigned int sStage[4][16 * 32];  // per-wave 16x64 bf16 staging

  const int tid = threadIdx.x;

  // One-time per persistent block: transposed bf16 weights WT[n][k], k packed 2/dword.
  for (int i = tid; i < 64 * 32; i += 128) {
    int n = i >> 5, kd = i & 31, k = kd << 1;
    sWm1[n * 32 + kd] = pk2(m_W1[k * 64 + n], m_W1[(k + 1) * 64 + n]);
    sWm2[n * 32 + kd] = pk2(m_W2[k * 64 + n], m_W2[(k + 1) * 64 + n]);
    sWe1[n * 32 + kd] = pk2(e_W1[k * 64 + n], e_W1[(k + 1) * 64 + n]);
    sWe2[n * 32 + kd] = pk2(e_W2[k * 64 + n], e_W2[(k + 1) * 64 + n]);
  }
  if (tid < 64) {
    float sc1 = m_g1[tid] * rsqrtf(m_v1[tid] + 1e-3f);
    float sc2 = m_g2[tid] * rsqrtf(m_v2[tid] + 1e-3f);
    sSc1[tid] = sc1; sSh1[tid] = m_b1[tid] - m_mu1[tid] * sc1;
    sSc2[tid] = sc2; sSh2[tid] = m_b2[tid] - m_mu2[tid] * sc2;
    sCm1[tid] = m_c1[tid]; sCm2[tid] = m_c2[tid];
    sCe1[tid] = e_c1[tid]; sCe2[tid] = e_c2[tid];
  }
  __syncthreads();

  const int lane = tid & 31;
  const int wave = tid >> 5;
  unsigned int* stage = sStage[wave];
  __bf16* stageB = (__bf16*)stage;
  const int colq = lane & 15;
  const int hi8 = (lane >= 16) ? 8 : 0;
  const int r = lane >> 1;
  const int c0 = (lane & 1) << 5;

  // Persistent loop: each wave processes one 16-edge tile per chunk.
  for (int chunk = blockIdx.x; chunk < NEDGES / 64; chunk += EDGE_BLOCKS) {
    const int tbase = chunk * 64 + wave * 16;

    // ---- message path: stage bn(node_feats[dst]) as bf16 (packed dword stores) ----
    {
      int nd = dst[tbase + r];
      const float* nf = node_feats + (long)nd * 64 + c0;
#pragma unroll
      for (int j = 0; j < 32; j += 2) {
        int c = c0 + j;
        stage[(r * 64 + c) >> 1] =
            pk2(nf[j] * sSc1[c] + sSh1[c], nf[j + 1] * sSc1[c + 1] + sSh1[c + 1]);
      }
    }
    ABf a0 = load_a(stage, 32, 0, lane);
    ABf a1 = load_a(stage, 32, 32, lane);
#pragma unroll
    for (int nt = 0; nt < 4; ++nt) {   // GEMM1: h = gelu(bn(x)@W1 + c1), then bn2 -> staged bf16
      v8f acc = {};
      acc = wmma_bf16(a0, load_b(sWm1, 32, nt * 16 + colq, 0, lane), acc);
      acc = wmma_bf16(a1, load_b(sWm1, 32, nt * 16 + colq, 32, lane), acc);
      int c = nt * 16 + colq;
      float bias = sCm1[c], s2 = sSc2[c], h2 = sSh2[c];
#pragma unroll
      for (int v = 0; v < 8; ++v) {
        float h = gelu_f(acc[v] + bias);
        stageB[(v + hi8) * 64 + c] = (__bf16)(h * s2 + h2);
      }
    }
    a0 = load_a(stage, 32, 0, lane);
    a1 = load_a(stage, 32, 32, lane);
    v8f msgD[4];
#pragma unroll
    for (int nt = 0; nt < 4; ++nt) {   // GEMM2: msg = gelu(. @ W2 + c2), keep in registers
      v8f acc = {};
      acc = wmma_bf16(a0, load_b(sWm2, 32, nt * 16 + colq, 0, lane), acc);
      acc = wmma_bf16(a1, load_b(sWm2, 32, nt * 16 + colq, 32, lane), acc);
      float bias = sCm2[nt * 16 + colq];
#pragma unroll
      for (int v = 0; v < 8; ++v) msgD[nt][v] = gelu_f(acc[v] + bias);
    }

    // ---- edge-gate path: er = gelu(gelu(edge_feats@We1+ce1)@We2+ce2) ----
    {
      const float* ef = edge_feats + (long)(tbase + r) * 64 + c0;
#pragma unroll
      for (int j = 0; j < 32; j += 2)
        stage[(r * 64 + c0 + j) >> 1] = pk2(ef[j], ef[j + 1]);
    }
    a0 = load_a(stage, 32, 0, lane);
    a1 = load_a(stage, 32, 32, lane);
#pragma unroll
    for (int nt = 0; nt < 4; ++nt) {
      v8f acc = {};
      acc = wmma_bf16(a0, load_b(sWe1, 32, nt * 16 + colq, 0, lane), acc);
      acc = wmma_bf16(a1, load_b(sWe1, 32, nt * 16 + colq, 32, lane), acc);
      int c = nt * 16 + colq;
      float bias = sCe1[c];
#pragma unroll
      for (int v = 0; v < 8; ++v)
        stageB[(v + hi8) * 64 + c] = (__bf16)gelu_f(acc[v] + bias);
    }
    a0 = load_a(stage, 32, 0, lane);
    a1 = load_a(stage, 32, 32, lane);
#pragma unroll
    for (int nt = 0; nt < 4; ++nt) {
      v8f acc = {};
      acc = wmma_bf16(a0, load_b(sWe2, 32, nt * 16 + colq, 0, lane), acc);
      acc = wmma_bf16(a1, load_b(sWe2, 32, nt * 16 + colq, 32, lane), acc);
      float bias = sCe2[nt * 16 + colq];
#pragma unroll
      for (int v = 0; v < 8; ++v) msgD[nt][v] *= gelu_f(acc[v] + bias);  // gate
    }

    // ---- scatter-add msg into segment sums ----
#pragma unroll
    for (int v = 0; v < 8; ++v) {
      int row = v + hi8;
      int sv = src[tbase + row];
      float* base = seg_sum + (long)sv * 64;
#pragma unroll
      for (int nt = 0; nt < 4; ++nt) atomic_add_f32(base + nt * 16 + colq, msgD[nt][v]);
    }
    if (lane < 16) atomic_add_f32(seg_cnt + src[tbase + lane], 1.0f);
  }
}

// ---------------- Node pipeline: mean -> concat -> BN -> GEMM(128->64) -> BN -> GEMM(64->64) ----------------
__global__ __launch_bounds__(128) void node_kernel(
    const float* __restrict__ node_feats,
    const float* __restrict__ seg_sum, const float* __restrict__ seg_cnt,
    const float* __restrict__ u_g1, const float* __restrict__ u_b1,
    const float* __restrict__ u_mu1, const float* __restrict__ u_v1,
    const float* __restrict__ u_W1, const float* __restrict__ u_c1,
    const float* __restrict__ u_g2, const float* __restrict__ u_b2,
    const float* __restrict__ u_mu2, const float* __restrict__ u_v2,
    const float* __restrict__ u_W2, const float* __restrict__ u_c2,
    float* __restrict__ out) {
  __shared__ unsigned int sW1[64 * 64];  // WT[n][k], K=128 -> 64 dwords/row
  __shared__ unsigned int sW2[64 * 32];  // WT[n][k], K=64  -> 32 dwords/row
  __shared__ float sSc1[128], sSh1[128], sSc2[64], sSh2[64], sC1[64], sC2[64];
  __shared__ unsigned int sStage[4][16 * 64];  // per-wave 16x128 bf16

  const int tid = threadIdx.x;
  for (int i = tid; i < 64 * 64; i += 128) {
    int n = i >> 6, kd = i & 63, k = kd << 1;
    sW1[n * 64 + kd] = pk2(u_W1[k * 64 + n], u_W1[(k + 1) * 64 + n]);
  }
  for (int i = tid; i < 64 * 32; i += 128) {
    int n = i >> 5, kd = i & 31, k = kd << 1;
    sW2[n * 32 + kd] = pk2(u_W2[k * 64 + n], u_W2[(k + 1) * 64 + n]);
  }
  if (tid < 128) {
    float sc = u_g1[tid] * rsqrtf(u_v1[tid] + 1e-3f);
    sSc1[tid] = sc; sSh1[tid] = u_b1[tid] - u_mu1[tid] * sc;
  }
  if (tid < 64) {
    float sc = u_g2[tid] * rsqrtf(u_v2[tid] + 1e-3f);
    sSc2[tid] = sc; sSh2[tid] = u_b2[tid] - u_mu2[tid] * sc;
    sC1[tid] = u_c1[tid]; sC2[tid] = u_c2[tid];
  }
  __syncthreads();

  const int lane = tid & 31;
  const int wave = tid >> 5;
  unsigned int* stage = sStage[wave];
  __bf16* stageB = (__bf16*)stage;
  const int colq = lane & 15;
  const int hi8 = (lane >= 16) ? 8 : 0;
  const int r = lane >> 1;

  for (int chunk = blockIdx.x; chunk < (NNODES + 63) / 64; chunk += NODE_BLOCKS) {
    const int tbase = chunk * 64 + wave * 16;
    if (tbase >= NNODES) continue;  // wave-uniform; EXEC stays all-ones for WMMA

    // Stage hcat = bn1([node_feats | seg_sum/max(cnt,1)]) as 16x128 bf16
    {
      int node = tbase + r; if (node >= NNODES) node = NNODES - 1;
      if ((lane & 1) == 0) {
        const float* nf = node_feats + (long)node * 64;
#pragma unroll
        for (int j = 0; j < 64; j += 2)
          stage[(r * 128 + j) >> 1] =
              pk2(nf[j] * sSc1[j] + sSh1[j], nf[j + 1] * sSc1[j + 1] + sSh1[j + 1]);
      } else {
        float inv = 1.0f / fmaxf(seg_cnt[node], 1.0f);
        const float* ss = seg_sum + (long)node * 64;
#pragma unroll
        for (int j = 0; j < 64; j += 2) {
          int c = 64 + j;
          stage[(r * 128 + c) >> 1] =
              pk2(ss[j] * inv * sSc1[c] + sSh1[c], ss[j + 1] * inv * sSc1[c + 1] + sSh1[c + 1]);
        }
      }
    }
    ABf a[4];
#pragma unroll
    for (int q = 0; q < 4; ++q) a[q] = load_a(stage, 64, q * 32, lane);
#pragma unroll
    for (int nt = 0; nt < 4; ++nt) {  // GEMM u1: [16x128] @ [128x64]
      v8f acc = {};
#pragma unroll
      for (int q = 0; q < 4; ++q)
        acc = wmma_bf16(a[q], load_b(sW1, 64, nt * 16 + colq, q * 32, lane), acc);
      int c = nt * 16 + colq;
      float bias = sC1[c], s2 = sSc2[c], h2 = sSh2[c];
#pragma unroll
      for (int v = 0; v < 8; ++v) {
        float h = gelu_f(acc[v] + bias);
        stageB[(v + hi8) * 64 + c] = (__bf16)(h * s2 + h2);  // restage 16x64
      }
    }
    ABf b0 = load_a(stage, 32, 0, lane);
    ABf b1 = load_a(stage, 32, 32, lane);
#pragma unroll
    for (int nt = 0; nt < 4; ++nt) {  // GEMM u2 + final gelu + guarded store
      v8f acc = {};
      acc = wmma_bf16(b0, load_b(sW2, 32, nt * 16 + colq, 0, lane), acc);
      acc = wmma_bf16(b1, load_b(sW2, 32, nt * 16 + colq, 32, lane), acc);
      int c = nt * 16 + colq;
      float bias = sC2[c];
#pragma unroll
      for (int v = 0; v < 8; ++v) {
        int node = tbase + v + hi8;
        if (node < NNODES) out[(long)node * 64 + c] = gelu_f(acc[v] + bias);
      }
    }
  }
}

extern "C" void kernel_launch(void* const* d_in, const int* in_sizes, int n_in,
                              void* d_out, int out_size, void* d_ws, size_t ws_size,
                              hipStream_t stream) {
  const float* node_feats = (const float*)d_in[0];
  const float* edge_feats = (const float*)d_in[1];
  const int*   src        = (const int*)d_in[2];
  const int*   dst        = (const int*)d_in[3];
  const float* m_g1 = (const float*)d_in[4];
  const float* m_b1 = (const float*)d_in[5];
  const float* m_mu1 = (const float*)d_in[6];
  const float* m_v1 = (const float*)d_in[7];
  const float* m_W1 = (const float*)d_in[8];
  const float* m_c1 = (const float*)d_in[9];
  const float* m_g2 = (const float*)d_in[10];
  const float* m_b2 = (const float*)d_in[11];
  const float* m_mu2 = (const float*)d_in[12];
  const float* m_v2 = (const float*)d_in[13];
  const float* m_W2 = (const float*)d_in[14];
  const float* m_c2 = (const float*)d_in[15];
  const float* e_W1 = (const float*)d_in[16];
  const float* e_c1 = (const float*)d_in[17];
  const float* e_W2 = (const float*)d_in[18];
  const float* e_c2 = (const float*)d_in[19];
  const float* u_g1 = (const float*)d_in[20];
  const float* u_b1 = (const float*)d_in[21];
  const float* u_mu1 = (const float*)d_in[22];
  const float* u_v1 = (const float*)d_in[23];
  const float* u_W1 = (const float*)d_in[24];
  const float* u_c1 = (const float*)d_in[25];
  const float* u_g2 = (const float*)d_in[26];
  const float* u_b2 = (const float*)d_in[27];
  const float* u_mu2 = (const float*)d_in[28];
  const float* u_v2 = (const float*)d_in[29];
  const float* u_W2 = (const float*)d_in[30];
  const float* u_c2 = (const float*)d_in[31];

  float* seg_sum = (float*)d_ws;                 // [N,64]
  float* seg_cnt = seg_sum + (long)NNODES * 64;  // [N]

  long zn = (long)NNODES * 65;
  zero_ws_kernel<<<(int)((zn + 255) / 256), 256, 0, stream>>>(seg_sum, zn);

  edge_kernel<<<EDGE_BLOCKS, 128, 0, stream>>>(
      node_feats, edge_feats, src, dst,
      m_g1, m_b1, m_mu1, m_v1, m_W1, m_c1,
      m_g2, m_b2, m_mu2, m_v2, m_W2, m_c2,
      e_W1, e_c1, e_W2, e_c2,
      seg_sum, seg_cnt);

  node_kernel<<<NODE_BLOCKS, 128, 0, stream>>>(
      node_feats, seg_sum, seg_cnt,
      u_g1, u_b1, u_mu1, u_v1, u_W1, u_c1,
      u_g2, u_b2, u_mu2, u_v2, u_W2, u_c2,
      (float*)d_out);
}
